// LTIRouter_48352741818709
// MI455X (gfx1250) — matrix operand order repaired and checked by reading, not compiled, not run on hardware
//
#include <hip/hip_runtime.h>
#include <hip/hip_bf16.h>

typedef __attribute__((ext_vector_type(16))) _Float16 v16h;
typedef __attribute__((ext_vector_type(8)))  _Float16 v8h;
typedef __attribute__((ext_vector_type(8)))  float    v8f;

#define D_TAPS 100
#define TLEN   2048
#define XPAD   112            // 7 blocks * 16 zero pad (causal head)
#define XBUF   (XPAD + TLEN)  // 2160 halfs, base stays 32B aligned

// ---------------------------------------------------------------------------
// Kernel 1: out = runoff  (vectorized copy, float4)
// ---------------------------------------------------------------------------
__global__ __launch_bounds__(256)
void lti_copy_kernel(const float4* __restrict__ in, float4* __restrict__ out,
                     long long n4) {
    long long i = (long long)blockIdx.x * blockDim.x + threadIdx.x;
    long long stride = (long long)gridDim.x * blockDim.x;
    for (; i < n4; i += stride) out[i] = in[i];
}

// ---------------------------------------------------------------------------
// Kernel 2: per-edge causal FIR via WMMA + atomic scatter
//   Y[i,j] = sum_q A_q[i,:] . B_q[:,j],  t = 16j + i
//   A_q[i,r] = h[16q+i-r],  B_q[r,j] = x[16(j-q)+r]
// q paired -> 4x v_wmma_f32_16x16x32_f16 per 16x16 output tile.
// ---------------------------------------------------------------------------
__global__ __launch_bounds__(256)
void lti_route_kernel(const float* __restrict__ runoff,
                      const int*   __restrict__ src,
                      const int*   __restrict__ dst,
                      const float* __restrict__ params,
                      float*       __restrict__ out,
                      int N) {
    __shared__ __align__(32) _Float16 xs[2][XBUF];   // f16 input, causal-padded
    __shared__ __align__(32) _Float16 As[4][16][32]; // packed Toeplitz tiles
    __shared__ float hs[128];                        // taps (100 used)

    const int e   = blockIdx.x;
    const int tid = threadIdx.x;
    const int se  = src[e];
    const int de  = dst[e];

    // --- 1) kernel taps: h[d] = r^d * (1-r)/(1-r^100), r = exp(-1/k) -------
    if (tid < D_TAPS) {
        float k    = 1.0f + 20.0f * params[e];
        float r    = __expf(-1.0f / k);
        float rd   = __expf(-(float)tid / k);
        float r100 = __expf(-100.0f / k);
        hs[tid] = rd * (1.0f - r) / (1.0f - r100);
    }
    __syncthreads();

    // --- 2) build A tiles: As[m][i][kk], d = 32m + 16*(kk>>4) + i - (kk&15)
    for (int idx = tid; idx < 4 * 16 * 32; idx += 256) {
        int m  = idx >> 9;
        int i  = (idx >> 5) & 15;
        int kk = idx & 31;
        int d  = 32 * m + 16 * (kk >> 4) + i - (kk & 15);
        float v = (d >= 0 && d < D_TAPS) ? hs[d] : 0.0f;
        As[m][i][kk] = (_Float16)v;
    }

    // --- 3) stage x for both batches (f32 -> f16), zero the causal pad -----
    for (int b = 0; b < 2; ++b) {
        const float* xg = runoff + ((size_t)b * N + se) * TLEN;
        for (int t = tid; t < XPAD; t += 256) xs[b][t] = (_Float16)0.0f;
        for (int t = tid; t < TLEN; t += 256) xs[b][XPAD + t] = (_Float16)xg[t];
    }
    __syncthreads();

    // --- 4) WMMA: 8 waves; waves 0-3 -> batch 0, waves 4-7 -> batch 1 ------
    const int wave = tid >> 5;
    const int lane = tid & 31;
    const int b    = wave >> 2;     // batch
    const int w4   = wave & 3;      // 2 j-tiles per wave
    const int n    = lane & 15;     // output column within tile (block idx)
    const int hi   = lane >> 4;     // lane-half selects K range

    // A operands: contiguous k order chosen here (B uses the same).
    //   lane<16 : K = {0..15};  lane>=16: K = {16..31}
    v16h a[4];
#pragma unroll
    for (int m = 0; m < 4; ++m) {
        v8h lo = *(const v8h*)&As[m][n][hi * 16];       // K base hi*16
        v8h up = *(const v8h*)&As[m][n][hi * 16 + 8];   // K base hi*16+8
        a[m] = __builtin_shufflevector(lo, up, 0, 1, 2, 3, 4, 5, 6, 7,
                                       8, 9, 10, 11, 12, 13, 14, 15);
    }

    float* outg = out + ((size_t)b * N + de) * TLEN;

#pragma unroll
    for (int jt = 0; jt < 2; ++jt) {
        const int j0 = (w4 * 2 + jt) * 16;   // base output block of this tile
        v8f acc = {};
#pragma unroll
        for (int m = 0; m < 4; ++m) {
            // B operand: 16-bit B 32x16 layout -> for this lane the 16 halfs
            // are contiguous x values starting at block (j0 + n - q), q=2m+hi.
            const int q = 2 * m + hi;
            v16h bv = *(const v16h*)&xs[b][XPAD + 16 * (j0 + n - q)];
            acc = __builtin_amdgcn_wmma_f32_16x16x32_f16(
                false, a[m], false, bv, (short)0, acc, false, false);
        }
        // scatter: row M = v + 8*hi, t = 16*(j0+n) + M  (8 contiguous floats)
        const int tbase = 16 * (j0 + n) + 8 * hi;
#pragma unroll
        for (int v = 0; v < 8; ++v)
            atomicAdd(&outg[tbase + v], acc[v]);
    }
}

// ---------------------------------------------------------------------------
extern "C" void kernel_launch(void* const* d_in, const int* in_sizes, int n_in,
                              void* d_out, int out_size, void* d_ws, size_t ws_size,
                              hipStream_t stream) {
    const float* runoff = (const float*)d_in[0];
    const int*   src    = (const int*)d_in[1];
    const int*   dst    = (const int*)d_in[2];
    const float* params = (const float*)d_in[3];
    float*       out    = (float*)d_out;

    const int E = in_sizes[1];
    const int B = 2;
    const int T = TLEN;
    const long long total = in_sizes[0];           // B*N*T
    const int N = (int)(total / ((long long)B * T));

    // out = runoff
    long long n4 = total / 4;
    int copyBlocks = (int)((n4 + 255) / 256);
    if (copyBlocks > 65535 * 8) copyBlocks = 65535 * 8;
    lti_copy_kernel<<<copyBlocks, 256, 0, stream>>>(
        (const float4*)runoff, (float4*)out, n4);

    // per-edge routed convolution + scatter
    lti_route_kernel<<<E, 256, 0, stream>>>(runoff, src, dst, params, out, N);
}
